// MolGraphEncoder_22239340658703
// MI455X (gfx1250) — matrix operand order repaired and checked by reading, not compiled, not run on hardware
//
#include <hip/hip_runtime.h>

#define H 128
#define TILE_M 64              // bonds per tile
#define TPB 5                  // tiles per block (E = 800000 = 2500 * 5 * 64)
#define KCH 4                  // 128 / 32 K-chunks
#define LDSF 132               // padded fp32 row stride (528B: 16B-aligned, staggers 64 banks)
#define TSZ (TILE_M * LDSF)    // floats per matrix tile in LDS
#define BUFSZ (3 * TSZ)        // floats per buffer (bond, xi, xj)
#define SMEM_BYTES (2 * BUFSZ * sizeof(float))   // ~198 KB < 320 KB/WGP
#define NCT_W 4                // column tiles per wave (wave covers a 64-col half)

typedef __attribute__((ext_vector_type(16))) __bf16 v16bf;
typedef __attribute__((ext_vector_type(8)))  float  v8f;

__device__ __forceinline__ unsigned short f2bf(float f) {
    unsigned int u = __float_as_uint(f);
    unsigned int r = u + 0x7FFFu + ((u >> 16) & 1u);   // round-to-nearest-even
    return (unsigned short)(r >> 16);
}

// CDNA5 async DMA: global -> LDS, 16B per lane. Same immediate offset is applied
// to both the LDS address (VDST vgpr) and the global address (VADDR vgpr pair).
__device__ __forceinline__ void async_cp16(unsigned lds, unsigned long long gaddr, int off) {
    asm volatile("global_load_async_to_lds_b128 %0, %1, off offset:%2"
                 :: "v"(lds), "v"(gaddr), "i"(off) : "memory");
}
__device__ __forceinline__ void wait_async0() {
    asm volatile("s_wait_asynccnt 0x0" ::: "memory");
}
// Low 32 bits of a flat pointer to LDS == byte offset within the workgroup allocation
__device__ __forceinline__ unsigned ldsoff(const void* p) {
    return (unsigned)(uintptr_t)p;
}

// Build one 16x32 bf16 A-fragment from the fp32 LDS tile (contiguous-k order, matched by B):
// lane holds row m, K = {koff..koff+15} of the 32-chunk.
__device__ __forceinline__ v16bf mk_afrag_f32(const float* p) {
    float4 a0 = *(const float4*)(p);
    float4 a1 = *(const float4*)(p + 4);
    float4 b0 = *(const float4*)(p + 8);
    float4 b1 = *(const float4*)(p + 12);
    v16bf r;
    r[0]  = (__bf16)a0.x; r[1]  = (__bf16)a0.y; r[2]  = (__bf16)a0.z; r[3]  = (__bf16)a0.w;
    r[4]  = (__bf16)a1.x; r[5]  = (__bf16)a1.y; r[6]  = (__bf16)a1.z; r[7]  = (__bf16)a1.w;
    r[8]  = (__bf16)b0.x; r[9]  = (__bf16)b0.y; r[10] = (__bf16)b0.z; r[11] = (__bf16)b0.w;
    r[12] = (__bf16)b1.x; r[13] = (__bf16)b1.y; r[14] = (__bf16)b1.z; r[15] = (__bf16)b1.w;
    return r;
}

// ---------------- kernel 0: zero the output accumulator ----------------
__global__ void mg_zero(float* __restrict__ out, int n) {
    int i = blockIdx.x * blockDim.x + threadIdx.x;
    if (i < n) out[i] = 0.0f;
}

// ---------------- kernel 1: weights fp32 -> bf16, fold gate biases ----------------
__global__ void mg_prep(const float* __restrict__ WU, const float* __restrict__ WV,
                        const float* __restrict__ WW, const float* __restrict__ WA,
                        const float* __restrict__ bU, const float* __restrict__ bV,
                        const float* __restrict__ bW,
                        unsigned short* __restrict__ wsW, float* __restrict__ bGate) {
    int id = blockIdx.x * blockDim.x + threadIdx.x;
    if (id < 4 * H * H) {
        int g = id >> 14;               // 0:WU 1:WV 2:WW 3:WA
        int rem = id & 16383;           // n*128 + k  (row-major W[n][k])
        const float* src = (g == 0) ? WU : (g == 1) ? WV : (g == 2) ? WW : WA;
        wsW[id] = f2bf(src[rem]);
    }
    if (id < H) bGate[id] = bU[id] + bV[id] + bW[id];
}

// ------- kernel 2: async double-buffered gather + 4 GEMMs + gate + segment-sum -------
__global__ __launch_bounds__(256) void mg_main(
    const float* __restrict__ atom, const float* __restrict__ bond,
    const int* __restrict__ idx_i, const int* __restrict__ idx_j,
    const int* __restrict__ seg_ids,
    const unsigned short* __restrict__ wsW,
    const float* __restrict__ bGate, const float* __restrict__ biasA,
    float* __restrict__ out)
{
    extern __shared__ float smem[];     // [2][3][TILE_M * LDSF] fp32, double-buffered

    const int t = threadIdx.x;
    const int tile0 = blockIdx.x * TPB;

    // wave decomposition: 8 waves = 4 row-stripes x 2 column-halves
    const int wid  = t >> 5;
    const int lane = t & 31;
    const int ws   = wid & 3;                 // row stripe (16 rows)
    const int ch   = wid >> 2;                // column half (64 cols)
    const int hi   = lane >> 4;
    const int m    = ws * 16 + (lane & 15);   // A-matrix row for this lane
    const int koffA = hi * 16;

    // issue one tile's gather as async global->LDS B128 copies (fp32)
    auto issue_tile = [&](int tile, int buf) {
        const int r  = t >> 2;                 // row 0..63
        const int hf = (t & 3) * 32;           // 32-float quarter row
        const int e  = tile * TILE_M + r;
        const int ri = idx_i[e], rj = idx_j[e];
        unsigned long long gb = (unsigned long long)(uintptr_t)(bond + (size_t)e  * H + hf);
        unsigned long long gi = (unsigned long long)(uintptr_t)(atom + (size_t)ri * H + hf);
        unsigned long long gj = (unsigned long long)(uintptr_t)(atom + (size_t)rj * H + hf);
        float* base = smem + buf * BUFSZ;
        unsigned lb = ldsoff(base + 0 * TSZ + r * LDSF + hf);
        unsigned li = ldsoff(base + 1 * TSZ + r * LDSF + hf);
        unsigned lj = ldsoff(base + 2 * TSZ + r * LDSF + hf);
        #pragma unroll
        for (int q = 0; q < 8; q++) {          // 8 x 16B = 32 floats
            async_cp16(lb, gb, q * 16);
            async_cp16(li, gi, q * 16);
            async_cp16(lj, gj, q * 16);
        }
    };

    issue_tile(tile0, 0);                      // prologue prefetch

    for (int tt = 0; tt < TPB; ++tt) {
        const int tile = tile0 + tt;
        const int buf  = tt & 1;

        wait_async0();                          // this wave's DMA into buf done
        __syncthreads();                        // everyone's DMA done -> buf readable

        // ---- build A-fragments (fp32 LDS -> bf16 VGPR), cached for whole tile ----
        const float* bufp = smem + buf * BUFSZ;
        v16bf aB[KCH], aI[KCH], aJ[KCH];
        #pragma unroll
        for (int kc = 0; kc < KCH; kc++) {
            const int o = m * LDSF + kc * 32 + koffA;
            aB[kc] = mk_afrag_f32(bufp + 0 * TSZ + o);
            aI[kc] = mk_afrag_f32(bufp + 1 * TSZ + o);
            aJ[kc] = mk_afrag_f32(bufp + 2 * TSZ + o);
        }
        __syncthreads();                        // all waves done reading buf

        if (tt + 1 < TPB) issue_tile(tile + 1, buf ^ 1);   // prefetch next tile

        // segment ids for the 8 rows this lane owns in the C/D layout
        const int e_base = tile * TILE_M + ws * 16 + hi * 8;
        int segs[8];
        #pragma unroll
        for (int i = 0; i < 8; i++) segs[i] = seg_ids[e_base + i];

        // ---- 4 column tiles x (4 K-chunks x 4 matrices) WMMAs per wave ----
        for (int ct = 0; ct < NCT_W; ct++) {
            const int n = ch * 64 + ct * 16 + (lane & 15);   // output column
            v8f accG = {};
            v8f accA = {};
            #pragma unroll
            for (int kc = 0; kc < KCH; kc++) {
                const int kb = kc * 32 + hi * 16;            // 16 contiguous k per lane
                const v16bf bu = *(const v16bf*)(wsW + 0 * H * H + n * H + kb);
                accG = __builtin_amdgcn_wmma_f32_16x16x32_bf16(false, aB[kc], false, bu, (short)0, accG, false, false);
                const v16bf bv = *(const v16bf*)(wsW + 1 * H * H + n * H + kb);
                accG = __builtin_amdgcn_wmma_f32_16x16x32_bf16(false, aI[kc], false, bv, (short)0, accG, false, false);
                const v16bf bw = *(const v16bf*)(wsW + 2 * H * H + n * H + kb);
                accG = __builtin_amdgcn_wmma_f32_16x16x32_bf16(false, aJ[kc], false, bw, (short)0, accG, false, false);
                const v16bf ba = *(const v16bf*)(wsW + 3 * H * H + n * H + kb);
                accA = __builtin_amdgcn_wmma_f32_16x16x32_bf16(false, aB[kc], false, ba, (short)0, accA, false, false);
            }
            const float bg  = bGate[n];
            const float bav = biasA[n];
            // run-length merged segment-sum atomics (seg_ids is sorted)
            float run = 0.0f;
            int   cur = segs[0];
            #pragma unroll
            for (int i = 0; i < 8; i++) {
                float g = 1.0f / (1.0f + __expf(-(accG[i] + bg)));
                float v = g * (accA[i] + bav);
                if (segs[i] != cur) {
                    atomicAdd(out + (size_t)cur * H + n, run);
                    run = 0.0f; cur = segs[i];
                }
                run += v;
            }
            atomicAdd(out + (size_t)cur * H + n, run);
        }
    }
}

// ---------------- kernel 3: divide by per-molecule bond count (binary search) ----------------
__global__ __launch_bounds__(128) void mg_final(const int* __restrict__ seg,
                                                float* __restrict__ out, int E) {
    const int s = blockIdx.x;
    int lo = 0, hi = E;
    while (lo < hi) { int mm = (lo + hi) >> 1; if (seg[mm] <  s) lo = mm + 1; else hi = mm; }
    const int start = lo;
    lo = 0; hi = E;
    while (lo < hi) { int mm = (lo + hi) >> 1; if (seg[mm] <= s) lo = mm + 1; else hi = mm; }
    const float inv = 1.0f / fmaxf((float)(lo - start), 1.0f);
    out[(size_t)s * H + threadIdx.x] *= inv;
}

extern "C" void kernel_launch(void* const* d_in, const int* in_sizes, int n_in,
                              void* d_out, int out_size, void* d_ws, size_t ws_size,
                              hipStream_t stream) {
    (void)n_in; (void)ws_size;
    const float* atom  = (const float*)d_in[0];
    const float* bond  = (const float*)d_in[1];
    const int*   idx_i = (const int*)d_in[2];
    const int*   idx_j = (const int*)d_in[3];
    const int*   seg   = (const int*)d_in[4];
    const float* WA = (const float*)d_in[5];
    const float* bA = (const float*)d_in[6];
    const float* WU = (const float*)d_in[7];
    const float* bU = (const float*)d_in[8];
    const float* WV = (const float*)d_in[9];
    const float* bV = (const float*)d_in[10];
    const float* WW = (const float*)d_in[11];
    const float* bW = (const float*)d_in[12];
    float* out = (float*)d_out;

    const int E      = in_sizes[2];          // 800000 = 2500 * TPB * TILE_M
    const int n_mols = out_size / H;         // 16384

    unsigned short* wsW   = (unsigned short*)d_ws;                       // 4*128*128 bf16
    float*          bGate = (float*)((char*)d_ws + 4 * H * H * sizeof(unsigned short));

    mg_zero<<<(out_size + 255) / 256, 256, 0, stream>>>(out, out_size);
    mg_prep<<<(4 * H * H + 255) / 256, 256, 0, stream>>>(WU, WV, WW, WA, bU, bV, bW, wsW, bGate);
    mg_main<<<E / (TILE_M * TPB), 256, SMEM_BYTES, stream>>>(atom, bond, idx_i, idx_j, seg,
                                                             wsW, bGate, bA, out);
    mg_final<<<n_mols, 128, 0, stream>>>(seg, out, E);
}
